// LSTM_2860448219669
// MI455X (gfx1250) — compile-verified, matrix operand order checked by reading, not run on hardware
//
#include <hip/hip_runtime.h>

// ---------------- problem constants ----------------
#define T_STEPS 2048
#define BATCH   64
#define DIN     128
#define HLAT    256
#define KTOT    (DIN + HLAT)      // 384
#define KCN     (KTOT / 32)       // 12 K-chunks of 32
#define NWG     8                 // persistent workgroups

typedef __bf16 bf16_t;
typedef bf16_t v16bf __attribute__((ext_vector_type(16)));
typedef float  v8f   __attribute__((ext_vector_type(8)));
typedef unsigned int v4u __attribute__((ext_vector_type(4)));

union Frag { v16bf v; v4u q[2]; };
union Pack16 { unsigned short h[16]; v4u q[2]; };

// ---------------- workspace layout (bytes) ----------------
static constexpr size_t BAR_OFF   = 0;                                   // 1 uint
static constexpr size_t HBUF_OFF  = 1024;                                // 2*64*256*2 = 65536
static constexpr size_t WFRAG_OFF = HBUF_OFF + 2ull * BATCH * HLAT * 2;  // 4*16*12*1024 = 786432
static constexpr size_t XBF_OFF   = ((WFRAG_OFF + 786432 + 4095) / 4096) * 4096;
// xbf: T*B*DIN bf16 = 32 MiB

__device__ __forceinline__ unsigned short f2bf(float f) {
  union { float f; unsigned u; } v; v.f = f;
  unsigned r = v.u + 0x7fffu + ((v.u >> 16) & 1u);   // round-to-nearest-even
  return (unsigned short)(r >> 16);
}
__device__ __forceinline__ float sigm(float x) { return 1.0f / (1.0f + __expf(-x)); }
__device__ __forceinline__ float tanh_fast(float x) { return 1.0f - 2.0f / (__expf(2.0f * x) + 1.0f); }

// ---------------- init: zero barrier/hbuf, build W B-fragments (bf16) ----------------
__global__ void lstm_init_w(const float* __restrict__ Wf, const float* __restrict__ Wi,
                            const float* __restrict__ Wo, const float* __restrict__ Wc,
                            unsigned char* __restrict__ ws) {
  int tid    = blockIdx.x * blockDim.x + threadIdx.x;
  int stride = gridDim.x * blockDim.x;
  if (tid == 0) *(unsigned*)(ws + BAR_OFF) = 0u;
  // zero both h double-buffers (h0 = 0)
  unsigned* hb = (unsigned*)(ws + HBUF_OFF);
  for (int i = tid; i < (2 * BATCH * HLAT * 2) / 4; i += stride) hb[i] = 0u;

  // B-fragment build: one lane-slot per thread. frag(gi, ht, kc): 32 lanes * 32 B.
  // B[k][n] = Wcat[gi*256 + ht*16 + n][k]; fused gate order f,i,g(=Wc),o.
  for (int idx = tid; idx < 4 * 16 * KCN * 32; idx += stride) {
    int lane = idx & 31;
    int kc   = (idx >> 5) % KCN;
    int ht   = (idx / (32 * KCN)) & 15;
    int gi   = idx / (32 * KCN * 16);
    const float* Wsrc = (gi == 0) ? Wf : (gi == 1) ? Wi : (gi == 2) ? Wc : Wo;
    int n   = lane & 15;
    int kb  = kc * 32 + ((lane >= 16) ? 8 : 0);
    int row = ht * 16 + n;                       // row in (256, 384), cols = [h | x]
    Pack16 p;
#pragma unroll
    for (int e = 0; e < 8; ++e) {
      p.h[e]     = f2bf(Wsrc[row * KTOT + kb + e]);
      p.h[8 + e] = f2bf(Wsrc[row * KTOT + kb + 16 + e]);
    }
    v4u* dst = (v4u*)(ws + WFRAG_OFF + (size_t)(((gi * 16 + ht) * KCN + kc) * 1024) + lane * 32);
    dst[0] = p.q[0];
    dst[1] = p.q[1];
  }
}

// ---------------- init: x -> bf16 ----------------
__global__ void lstm_cvt_x(const float* __restrict__ x, unsigned short* __restrict__ xbf, int n) {
  int i = blockIdx.x * blockDim.x + threadIdx.x;
  if (i < n) xbf[i] = f2bf(x[i]);
}

// ---------------- persistent recurrent kernel ----------------
// 8 WGs x 256 threads (8 waves). Wave = (batch-tile mt, hidden-tile ht); owns all 4 gates.
__launch_bounds__(256, 1)
__global__ void lstm_persist(const float* __restrict__ bfp, const float* __restrict__ bip,
                             const float* __restrict__ bop, const float* __restrict__ bcp,
                             float* __restrict__ out, unsigned char* __restrict__ ws) {
  const int wave = threadIdx.x >> 5;
  const int lane = threadIdx.x & 31;
  const int mt   = wave & 3;                  // 4 batch tiles of 16
  const int ht   = blockIdx.x * 2 + (wave >> 2);  // 16 hidden tiles of 16
  const int m0   = mt * 16;
  const int laneN    = lane & 15;
  const int laneHalf = lane >> 4;
  const int laneM = m0 + laneN;               // A-matrix row (batch)
  const int j     = ht * 16 + laneN;          // C/D column (hidden unit)

  unsigned*             bar   = (unsigned*)(ws + BAR_OFF);
  unsigned short*       hbuf  = (unsigned short*)(ws + HBUF_OFF);   // [2][64][256] bf16
  const unsigned char*  wfrag = ws + WFRAG_OFF;
  const unsigned short* xbf   = (const unsigned short*)(ws + XBF_OFF);

  // ---- hoist all weight B-fragments into VGPRs (loop-invariant, 384 VGPRs) ----
  Frag wB[4][KCN];
#pragma unroll
  for (int gi = 0; gi < 4; ++gi)
#pragma unroll
    for (int kc = 0; kc < KCN; ++kc) {
      const v4u* p = (const v4u*)(wfrag + (size_t)(((gi * 16 + ht) * KCN + kc) * 1024) + lane * 32);
      wB[gi][kc].q[0] = p[0];
      wB[gi][kc].q[1] = p[1];
    }

  const float bfv = bfp[j], biv = bip[j], bgv = bcp[j], bov = bop[j];

  float* hT = out;
  float* hs = out + (size_t)BATCH * HLAT;
  float* cT = hs + (size_t)T_STEPS * BATCH * HLAT;
  float* cs = cT + (size_t)BATCH * HLAT;

  v8f cstate = {};   // c0 = 0

  for (int t = 0; t < T_STEPS; ++t) {
    const unsigned short* hrow = hbuf + (t & 1) * (BATCH * HLAT);
    const unsigned short* xrow = xbf + (size_t)t * BATCH * DIN;

    v8f zf, zi, zg, zo;
#pragma unroll
    for (int r = 0; r < 8; ++r) { zf[r] = bfv; zi[r] = biv; zg[r] = bgv; zo[r] = bov; }

#pragma unroll
    for (int kc = 0; kc < KCN; ++kc) {
      Frag a;
      if (kc < 8) {  // K 0..255 : recurrent h (bf16 double buffer)
        const v4u* pa = (const v4u*)(hrow + laneM * HLAT + kc * 32 + laneHalf * 8);
        a.q[0] = pa[0];
        a.q[1] = pa[2];   // K + 16
      } else {       // K 256..383 : x_t (pre-converted bf16)
        const v4u* pa = (const v4u*)(xrow + laneM * DIN + (kc - 8) * 32 + laneHalf * 8);
        a.q[0] = pa[0];
        a.q[1] = pa[2];
      }
      zf = __builtin_amdgcn_wmma_f32_16x16x32_bf16(false, a.v, false, wB[0][kc].v, (short)0, zf, false, false);
      zi = __builtin_amdgcn_wmma_f32_16x16x32_bf16(false, a.v, false, wB[1][kc].v, (short)0, zi, false, false);
      zg = __builtin_amdgcn_wmma_f32_16x16x32_bf16(false, a.v, false, wB[2][kc].v, (short)0, zg, false, false);
      zo = __builtin_amdgcn_wmma_f32_16x16x32_bf16(false, a.v, false, wB[3][kc].v, (short)0, zo, false, false);
    }

    // ---- gates, c/h update, outputs (hs/cs are stream-once -> non-temporal,
    //      keeps x/weights resident in the 192MB L2 across all 2048 steps) ----
    unsigned short* hw = hbuf + ((t + 1) & 1) * (BATCH * HLAT);
#pragma unroll
    for (int r = 0; r < 8; ++r) {
      int   m  = m0 + r + laneHalf * 8;       // C/D row -> batch index
      float f  = sigm(zf[r]);
      float i_ = sigm(zi[r]);
      float g  = tanh_fast(zg[r]);
      float o  = sigm(zo[r]);
      float cn = f * cstate[r] + i_ * g;
      cstate[r] = cn;
      float hn = o * tanh_fast(cn);
      size_t oidx = ((size_t)t * BATCH + m) * HLAT + j;
      __builtin_nontemporal_store(hn, &hs[oidx]);
      __builtin_nontemporal_store(cn, &cs[oidx]);
      if (t == T_STEPS - 1) { hT[m * HLAT + j] = hn; cT[m * HLAT + j] = cn; }
      hw[m * HLAT + j] = f2bf(hn);            // next step's A operand
    }

    // prefetch next timestep's x slice for this wave (hidden under the barrier)
    if (t + 1 < T_STEPS) {
      const unsigned short* xnext = xbf + (size_t)(t + 1) * BATCH * DIN + laneM * DIN;
      __builtin_prefetch(xnext, 0, 3);
      __builtin_prefetch(xnext + 64, 0, 3);
    }

    // ---- grid barrier (release/acquire, agent scope; per-wave acquire spin
    //      so every CU's L0 is invalidated before reading remote h) ----
    __threadfence();
    __syncthreads();
    if (threadIdx.x == 0)
      __hip_atomic_fetch_add(bar, 1u, __ATOMIC_RELEASE, __HIP_MEMORY_SCOPE_AGENT);
    unsigned target = (unsigned)(t + 1) * (unsigned)gridDim.x;
    if (lane == 0) {
      while (__hip_atomic_load(bar, __ATOMIC_ACQUIRE, __HIP_MEMORY_SCOPE_AGENT) < target)
        __builtin_amdgcn_s_sleep(8);
    }
    __syncthreads();
  }
}

// ---------------- launch ----------------
extern "C" void kernel_launch(void* const* d_in, const int* in_sizes, int n_in,
                              void* d_out, int out_size, void* d_ws, size_t ws_size,
                              hipStream_t stream) {
  (void)in_sizes; (void)n_in; (void)out_size; (void)ws_size;
  const float* x  = (const float*)d_in[0];
  const float* Wf = (const float*)d_in[1];
  const float* bf = (const float*)d_in[2];
  const float* Wi = (const float*)d_in[3];
  const float* bi = (const float*)d_in[4];
  const float* Wo = (const float*)d_in[5];
  const float* bo = (const float*)d_in[6];
  const float* Wc = (const float*)d_in[7];
  const float* bc = (const float*)d_in[8];
  unsigned char* ws = (unsigned char*)d_ws;
  float* out = (float*)d_out;

  lstm_init_w<<<64, 256, 0, stream>>>(Wf, Wi, Wo, Wc, ws);

  const int nx = T_STEPS * BATCH * DIN;
  lstm_cvt_x<<<(nx + 255) / 256, 256, 0, stream>>>(
      x, (unsigned short*)(ws + XBF_OFF), nx);

  lstm_persist<<<NWG, 256, 0, stream>>>(bf, bi, bo, bc, out, ws);
}
